// SGNP_26010321944693
// MI455X (gfx1250) — compile-verified, matrix operand order checked by reading, not run on hardware
//
#include <hip/hip_runtime.h>
#include <hip/hip_bf16.h>
#include <math.h>

#define B_    8
#define NC    2048
#define NT    2048
#define D_    2
#define H_    64
#define E_    4
#define KNN   32
#define NBLK  6
#define BNC   (B_*NC)          // 16384 ctx nodes
#define BNT   (B_*NT)          // 16384 test nodes
#define NNODE (BNC+BNT)        // 32768 total nodes
#define DIN   7                // E + D + Fd

typedef __attribute__((ext_vector_type(16))) _Float16 v16h;
typedef __attribute__((ext_vector_type(8)))  float    v8f;

// ---------- WMMA fragment helpers (CDNA5 16x16x32 f16 layout) ----------
// half j (0..15), hi = lane>=16  ->  K offset within a 32-wide chunk
__device__ __forceinline__ int kmap(int j, int hi) {
    return ((j >> 3) << 4) + (hi << 3) + (j & 7);
}

__device__ __forceinline__ v8f wmma_f16(v16h a, v16h b, v8f c) {
    return __builtin_amdgcn_wmma_f32_16x16x32_f16(
        /*neg_a=*/false, a, /*neg_b=*/false, b,
        /*c_mod=*/(short)0, c, /*reuse_a=*/false, /*reuse_b=*/false);
}

// contiguous fragment load (32 bytes/lane) from fragment-major f16 buffer
__device__ __forceinline__ v16h load_frag(const _Float16* __restrict__ base,
                                          int fragIdx, int lane) {
    return *(const v16h*)(base + ((size_t)fragIdx * 32 + lane) * 16);
}
__device__ __forceinline__ v16h load_frag_lds(const _Float16* base,
                                              int fragIdx, int lane) {
    return *(const v16h*)(base + ((size_t)fragIdx * 32 + lane) * 16);
}

// fast, branch-free activations (v_exp_f32 / v_rcp_f32 / v_rsq_f32 paths)
__device__ __forceinline__ float gelu_f(float x) {
    float u = 0.7978845608028654f * (x + 0.044715f * x * x * x);
    float e = __expf(2.0f * u);                       // exp overflow -> +inf ok
    float t = 1.0f - 2.0f * __builtin_amdgcn_rcpf(e + 1.0f);
    return 0.5f * x * (1.0f + t);
}

// store a C fragment (+bias, opt. GELU) into an A-fragment-layout f16 LDS buf.
// Output column 'col' of this layer is K index 'col' of the next layer.
__device__ __forceinline__ void storeC_frag_lds(_Float16* X, int nt, v8f c,
                                                const float* __restrict__ bias,
                                                int m, int hi, bool do_gelu) {
    int kc   = nt >> 1;
    int hi2  = (m >> 3) & 1;
    int j2   = ((nt & 1) << 3) | (m & 7);
    float bb = bias[nt * 16 + m];
#pragma unroll
    for (int r = 0; r < 8; ++r) {
        int row = r + hi * 8;
        int lane2 = row + hi2 * 16;
        float v = c[r] + bb;
        if (do_gelu) v = gelu_f(v);
        X[(kc * 32 + lane2) * 16 + j2] = (_Float16)v;
    }
}

// =====================================================================
// Kernel 0: weight prep — convert f32 KxN weights to f16 fragment-major
// layout [kc*nNt+nt][lane][16], zero-padded beyond Krows. 1 wave/fragment.
// =====================================================================
__global__ void __launch_bounds__(32) SGNP_wprep_kernel(
    const float* __restrict__ W, int N, int Krows, int nNt,
    _Float16* __restrict__ out)
{
    int idx = blockIdx.x;                 // kc*nNt + nt
    int kc = idx / nNt, nt = idx % nNt;
    int lane = threadIdx.x & 31;
    int n = nt * 16 + (lane & 15), hi = (lane >> 4) & 1;
    _Float16* o = out + ((size_t)idx * 32 + lane) * 16;
#pragma unroll
    for (int j = 0; j < 16; ++j) {
        int k = kc * 32 + kmap(j, hi);
        o[j] = (_Float16)((k < Krows) ? W[k * N + n] : 0.0f);
    }
}

// =====================================================================
// Kernel 1: encoder MLP 7->256->128->64 (+GELU,+GELU, layernorm)
// writes nodes (f32) and nodes16 (f16, A-fragment layout). 2 waves/block.
// =====================================================================
__global__ void __launch_bounds__(64) SGNP_enc_kernel(
    const float* __restrict__ s_ctx, const float* __restrict__ f_ctx,
    const float* __restrict__ s_test, const float* __restrict__ emb,
    const _Float16* __restrict__ fW0, const float* __restrict__ b0,
    const _Float16* __restrict__ fW1, const float* __restrict__ b1,
    const _Float16* __restrict__ fW2, const float* __restrict__ b2,
    const float* __restrict__ lns, const float* __restrict__ lnb,
    float* __restrict__ nodes, _Float16* __restrict__ nodes16)
{
    __shared__ __align__(32) _Float16 sA[2][8][32][16];  // 256-wide, frag layout
    __shared__ __align__(32) _Float16 sB[2][4][32][16];  // 128-wide, frag layout
    __shared__ float sF[2][16][64];                      // 64-wide, row major

    int w = threadIdx.x >> 5, lane = threadIdx.x & 31;
    int m = lane & 15, hi = (lane >> 4) & 1;
    int tile = blockIdx.x * 2 + w;
    int base = tile * 16;
    int node = base + m;

    // ---- input features, branch-free A0 fragment (K padded 7 -> 32) ----
    bool is_test = node >= BNC;
    const float* er = emb + (is_test ? 0 : 1) * E_;
    float sx, sy, fv;
    if (is_test) {
        int t = node - BNC;
        sx = s_test[t * 2 + 0]; sy = s_test[t * 2 + 1]; fv = 0.0f;
    } else {
        sx = s_ctx[node * 2 + 0]; sy = s_ctx[node * 2 + 1]; fv = f_ctx[node];
    }
    float vals[8] = { er[0], er[1], er[2], er[3], sx, sy, fv, 0.0f };
    v16h a0;
#pragma unroll
    for (int j = 0; j < 8; ++j) a0[j] = (_Float16)(hi ? 0.0f : vals[j]);
#pragma unroll
    for (int j = 8; j < 16; ++j) a0[j] = (_Float16)0.0f;

    // ---- layer 0: 7(pad32) -> 256, GELU ----
    for (int nt = 0; nt < 16; ++nt) {
        v16h b = load_frag(fW0, nt, lane);        // nKc=1
        v8f c = {};
        c = wmma_f16(a0, b, c);
        storeC_frag_lds(&sA[w][0][0][0], nt, c, b0, m, hi, true);
    }
    __syncthreads();

    // ---- layer 1: 256 -> 128, GELU ----
    for (int nt = 0; nt < 8; ++nt) {
        v8f c = {};
        for (int kc = 0; kc < 8; ++kc) {
            v16h a = load_frag_lds(&sA[w][0][0][0], kc, lane);
            v16h b = load_frag(fW1, kc * 8 + nt, lane);
            c = wmma_f16(a, b, c);
        }
        storeC_frag_lds(&sB[w][0][0][0], nt, c, b1, m, hi, true);
    }
    __syncthreads();

    // ---- layer 2: 128 -> 64 (no activation) ----
    for (int nt = 0; nt < 4; ++nt) {
        v8f c = {};
        for (int kc = 0; kc < 4; ++kc) {
            v16h a = load_frag_lds(&sB[w][0][0][0], kc, lane);
            v16h b = load_frag(fW2, kc * 4 + nt, lane);
            c = wmma_f16(a, b, c);
        }
#pragma unroll
        for (int r = 0; r < 8; ++r) {
            int row = r + hi * 8, col = nt * 16 + m;
            sF[w][row][col] = c[r] + b2[col];
        }
    }
    __syncthreads();

    // ---- layernorm over 64; lanes m and m+16 share row m ----
    float s1 = 0.f, s2 = 0.f;
#pragma unroll
    for (int i = 0; i < 32; ++i) {
        float v = sF[w][m][hi * 32 + i];
        s1 += v; s2 += v * v;
    }
    s1 += __shfl_xor(s1, 16);
    s2 += __shfl_xor(s2, 16);
    float mean = s1 * (1.0f / 64.0f);
    float var  = s2 * (1.0f / 64.0f) - mean * mean;
    float rstd = __builtin_amdgcn_rsqf(var + 1e-6f);
#pragma unroll
    for (int i = 0; i < 32; ++i) {
        int f = hi * 32 + i;
        float v = sF[w][m][f];
        float y = (v - mean) * rstd * lns[f] + lnb[f];
        nodes[(size_t)(base + m) * H_ + f] = y;
        int hi2 = (i >> 3) & 1, j2 = ((i >> 4) << 3) | (i & 7);
        int lane2 = m + hi2 * 16;
        nodes16[((size_t)(tile * 2 + hi) * 32 + lane2) * 16 + j2] = (_Float16)y;
    }
}

// =====================================================================
// Kernel 2: exact KNN (K=32) per query, insertion list over LDS keys
// =====================================================================
__global__ void __launch_bounds__(256) SGNP_knn_kernel(
    const float* __restrict__ s_ctx, const float* __restrict__ s_test,
    const int* __restrict__ vlen,
    int* __restrict__ send, float* __restrict__ emask,
    float* __restrict__ edges)
{
    __shared__ float kx[NC], ky[NC];
    int blk = blockIdx.x;
    bool test_set = blk >= 64;
    int sblk = test_set ? blk - 64 : blk;
    int b = sblk >> 3;
    int local = (sblk & 7) * 256 + threadIdx.x;
    int vl = vlen[b];
    const float INF = __builtin_inff();

    for (int j = threadIdx.x; j < NC; j += 256) {
        bool valid = j < vl;
        kx[j] = valid ? s_ctx[(b * NC + j) * 2 + 0] : INF;
        ky[j] = valid ? s_ctx[(b * NC + j) * 2 + 1] : INF;
    }
    __syncthreads();

    float qx, qy; int qrow;
    if (test_set) {
        qx = s_test[(b * NT + local) * 2 + 0];
        qy = s_test[(b * NT + local) * 2 + 1];
        qrow = BNC + b * NT + local;
    } else {
        qx = s_ctx[(b * NC + local) * 2 + 0];
        qy = s_ctx[(b * NC + local) * 2 + 1];
        qrow = b * NC + local;
    }

    float bd[KNN]; int bi[KNN];
    for (int k = 0; k < KNN; ++k) { bd[k] = INF; bi[k] = 0; }

    for (int j = 0; j < NC; ++j) {
        float dx = kx[j] - qx, dy = ky[j] - qy;
        float d2 = dx * dx + dy * dy;
        if (d2 < bd[KNN - 1]) {
            int p = KNN - 1;
            while (p > 0 && bd[p - 1] > d2) {
                bd[p] = bd[p - 1]; bi[p] = bi[p - 1]; --p;
            }
            bd[p] = d2; bi[p] = j;
        }
    }

    for (int k = 0; k < KNN; ++k) {
        int idx = bi[k];
        float dx = kx[idx] - qx, dy = ky[idx] - qy;
        bool fin = __builtin_isfinite(dx) && __builtin_isfinite(dy);
        size_t e = (size_t)qrow * KNN + k;
        send[e]  = b * NC + idx;
        emask[e] = fin ? 1.0f : 0.0f;
        edges[e * 2 + 0] = fin ? dx : 0.0f;
        edges[e * 2 + 1] = fin ? dy : 0.0f;
    }
}

// =====================================================================
// Kernel 3a: h = nodes @ W (WMMA from f16 fragments), hr = h . a_r
// =====================================================================
__global__ void __launch_bounds__(64) SGNP_gat_h_kernel(
    const _Float16* __restrict__ nodes16, const _Float16* __restrict__ fgW,
    const float* __restrict__ gar,
    float* __restrict__ h, float* __restrict__ hr)
{
    __shared__ float sH[2][16][65];
    int w = threadIdx.x >> 5, lane = threadIdx.x & 31;
    int m = lane & 15, hi = (lane >> 4) & 1;
    int tile = blockIdx.x * 2 + w;
    int base = tile * 16;

    for (int nt = 0; nt < 4; ++nt) {
        v8f c = {};
        for (int kc = 0; kc < 2; ++kc) {
            v16h a = load_frag(nodes16, tile * 2 + kc, lane);
            v16h b = load_frag(fgW, kc * 4 + nt, lane);
            c = wmma_f16(a, b, c);
        }
#pragma unroll
        for (int r = 0; r < 8; ++r) {
            int row = r + hi * 8, col = nt * 16 + m;
            float v = c[r];
            h[(size_t)(base + row) * H_ + col] = v;
            sH[w][row][col] = v;
        }
    }
    __syncthreads();

    float p = 0.f;
#pragma unroll
    for (int i = 0; i < 32; ++i) {
        int f = hi * 32 + i;
        p += sH[w][m][f] * gar[f];
    }
    p += __shfl_xor(p, 16);
    if (hi == 0) hr[base + m] = p;
}

// =====================================================================
// Kernel 3b: attention + aggregation + residual layernorm.
// 4 waves/block, one node/wave; lane = neighbor k. Alphas via shfl.
// Writes nodes (f32) and nodes16 (f16 fragment shadow).
// =====================================================================
__global__ void __launch_bounds__(128) SGNP_gat_agg_kernel(
    float* __restrict__ nodes, _Float16* __restrict__ nodes16,
    const float* __restrict__ h, const float* __restrict__ hr,
    const int* __restrict__ send, const float* __restrict__ emask,
    const float* __restrict__ edges,
    const float* __restrict__ gas, const float* __restrict__ gae,
    const float* __restrict__ lns, const float* __restrict__ lnb)
{
    __shared__ float hs[4][32][65];   // padded: no k-striped bank conflicts
    int w = threadIdx.x >> 5, lane = threadIdx.x & 31;
    int n = blockIdx.x * 4 + w;
    int k = lane;

    size_t e = (size_t)n * KNN + k;
    int s = send[e];
    float mk = emask[e];
    float e0 = edges[e * 2 + 0], e1 = edges[e * 2 + 1];

    // gather neighbor row (L2-resident) into LDS + dot with a_s
    const float4* h4 = (const float4*)(h + (size_t)s * H_);
    float das = 0.f;
#pragma unroll
    for (int f4 = 0; f4 < H_ / 4; ++f4) {
        float4 v = h4[f4];
        int f = f4 * 4;
        hs[w][k][f + 0] = v.x; hs[w][k][f + 1] = v.y;
        hs[w][k][f + 2] = v.z; hs[w][k][f + 3] = v.w;
        das += v.x * gas[f] + v.y * gas[f + 1] + v.z * gas[f + 2] + v.w * gas[f + 3];
    }

    float lg = hr[n] + das + e0 * gae[0] + e1 * gae[1];
    lg = (lg > 0.f) ? lg : 0.2f * lg;             // leaky_relu(0.2)
    lg = (mk > 0.5f) ? lg : -1.0e9f;              // NEG mask

    float mx = lg;
#pragma unroll
    for (int off = 16; off >= 1; off >>= 1) mx = fmaxf(mx, __shfl_xor(mx, off));
    float ex = __expf(lg - mx);
    float sm = ex;
#pragma unroll
    for (int off = 16; off >= 1; off >>= 1) sm += __shfl_xor(sm, off);
    float al = (mk > 0.5f) ? ex * __builtin_amdgcn_rcpf(sm) : 0.0f;
    __syncthreads();

    // aggregation: lane owns features lane and lane+32; alpha via shfl
    float acc0 = 0.f, acc1 = 0.f;
#pragma unroll
    for (int kk = 0; kk < 32; ++kk) {
        float a = __shfl(al, kk);
        acc0 += a * hs[w][kk][lane];
        acc1 += a * hs[w][kk][lane + 32];
    }
    float x0 = nodes[(size_t)n * H_ + lane]      + acc0;
    float x1 = nodes[(size_t)n * H_ + lane + 32] + acc1;
    float s1 = x0 + x1, s2 = x0 * x0 + x1 * x1;
#pragma unroll
    for (int off = 16; off >= 1; off >>= 1) {
        s1 += __shfl_xor(s1, off);
        s2 += __shfl_xor(s2, off);
    }
    float mean = s1 * (1.0f / 64.0f);
    float var  = s2 * (1.0f / 64.0f) - mean * mean;
    float rstd = __builtin_amdgcn_rsqf(var + 1e-6f);
    float y0 = (x0 - mean) * rstd * lns[lane]      + lnb[lane];
    float y1 = (x1 - mean) * rstd * lns[lane + 32] + lnb[lane + 32];
    nodes[(size_t)n * H_ + lane]      = y0;
    nodes[(size_t)n * H_ + lane + 32] = y1;

    // f16 fragment shadow for next block's WMMA
    int t = n >> 4, row = n & 15;
    int hi2 = (lane >> 3) & 1, j2 = ((lane >> 4) << 3) | (lane & 7);
    int lane2 = row + hi2 * 16;
    nodes16[((size_t)(t * 2 + 0) * 32 + lane2) * 16 + j2] = (_Float16)y0;
    nodes16[((size_t)(t * 2 + 1) * 32 + lane2) * 16 + j2] = (_Float16)y1;
}

// =====================================================================
// Kernel 4: head MLP 64->256->64->2, loc + softplus(scale)
// =====================================================================
__global__ void __launch_bounds__(64) SGNP_head_kernel(
    const _Float16* __restrict__ nodes16,
    const _Float16* __restrict__ fW0, const float* __restrict__ b0,
    const _Float16* __restrict__ fW1, const float* __restrict__ b1,
    const float* __restrict__ W2, const float* __restrict__ b2,
    float* __restrict__ out)
{
    __shared__ __align__(32) _Float16 sA[2][8][32][16];  // 256-wide frag
    __shared__ float sF[2][16][64];
    int w = threadIdx.x >> 5, lane = threadIdx.x & 31;
    int m = lane & 15, hi = (lane >> 4) & 1;
    int tile = blockIdx.x * 2 + w;
    int gtile = (BNC / 16) + tile;        // test-node tiles

    // layer 0: 64 -> 256, GELU
    for (int nt = 0; nt < 16; ++nt) {
        v8f c = {};
        for (int kc = 0; kc < 2; ++kc) {
            v16h a = load_frag(nodes16, gtile * 2 + kc, lane);
            v16h b = load_frag(fW0, kc * 16 + nt, lane);
            c = wmma_f16(a, b, c);
        }
        storeC_frag_lds(&sA[w][0][0][0], nt, c, b0, m, hi, true);
    }
    __syncthreads();

    // layer 1: 256 -> 64, GELU (row-major store for scalar layer 2)
    for (int nt = 0; nt < 4; ++nt) {
        v8f c = {};
        for (int kc = 0; kc < 8; ++kc) {
            v16h a = load_frag_lds(&sA[w][0][0][0], kc, lane);
            v16h b = load_frag(fW1, kc * 4 + nt, lane);
            c = wmma_f16(a, b, c);
        }
#pragma unroll
        for (int r = 0; r < 8; ++r) {
            int row = r + hi * 8, col = nt * 16 + m;
            sF[w][row][col] = gelu_f(c[r] + b1[col]);
        }
    }
    __syncthreads();

    // layer 2: 64 -> 2 scalar dots; loc + softplus(scale)
    float p0 = 0.f, p1 = 0.f;
#pragma unroll
    for (int i = 0; i < 32; ++i) {
        int f = hi * 32 + i;
        float v = sF[w][m][f];
        p0 += v * W2[f * 2 + 0];
        p1 += v * W2[f * 2 + 1];
    }
    p0 += __shfl_xor(p0, 16);
    p1 += __shfl_xor(p1, 16);
    if (hi == 0) {
        int t = tile * 16 + m;
        float o0 = p0 + b2[0], o1 = p1 + b2[1];
        out[(size_t)t * 2 + 0] = o0;
        out[(size_t)t * 2 + 1] = (o1 > 20.0f) ? o1 : __logf(1.0f + __expf(o1));
    }
}

// =====================================================================
extern "C" void kernel_launch(void* const* d_in, const int* in_sizes, int n_in,
                              void* d_out, int out_size, void* d_ws, size_t ws_size,
                              hipStream_t stream)
{
    (void)in_sizes; (void)n_in; (void)out_size; (void)ws_size;

    const float* s_ctx = (const float*)d_in[0];
    const float* f_ctx = (const float*)d_in[1];
    const float* s_test= (const float*)d_in[2];
    const int*   vlen  = (const int*)  d_in[3];
    const float* emb   = (const float*)d_in[4];
    const float* eaW0  = (const float*)d_in[5];
    const float* eab0  = (const float*)d_in[6];
    const float* eaW1  = (const float*)d_in[7];
    const float* eab1  = (const float*)d_in[8];
    const float* eaW2  = (const float*)d_in[9];
    const float* eab2  = (const float*)d_in[10];
    const float* lns   = (const float*)d_in[11];
    const float* lnb   = (const float*)d_in[12];
    const float* gW    = (const float*)d_in[13];
    const float* gas   = (const float*)d_in[14];
    const float* gar   = (const float*)d_in[15];
    const float* gae   = (const float*)d_in[16];
    const float* glns  = (const float*)d_in[17];
    const float* glnb  = (const float*)d_in[18];
    const float* hW0   = (const float*)d_in[19];
    const float* hb0   = (const float*)d_in[20];
    const float* hW1   = (const float*)d_in[21];
    const float* hb1   = (const float*)d_in[22];
    const float* hW2   = (const float*)d_in[23];
    const float* hb2   = (const float*)d_in[24];
    float* out = (float*)d_out;

    // ---- workspace layout ----
    float* nodes = (float*)d_ws;                               // 32768*64 f32
    float* hbuf  = nodes + (size_t)NNODE * H_;
    float* hr    = hbuf  + (size_t)NNODE * H_;
    float* emask = hr    + NNODE;
    float* edges = emask + (size_t)NNODE * KNN;
    int*   send  = (int*)(edges + (size_t)NNODE * KNN * 2);
    _Float16* nodes16 = (_Float16*)(send + (size_t)NNODE * KNN);   // frag shadow
    _Float16* fW0enc = nodes16 + (size_t)NNODE * H_;
    _Float16* fW1enc = fW0enc + 16 * 512;   // 1x16 frags
    _Float16* fW2enc = fW1enc + 64 * 512;   // 8x8
    _Float16* fgW    = fW2enc + 16 * 512;   // 4x4
    _Float16* fhW0   = fgW    + 48 * 512;   // 6 blocks x (2x4)
    _Float16* fhW1   = fhW0   + 32 * 512;   // 2x16

    // ---- weight prep: f32 -> f16 fragment-major ----
    SGNP_wprep_kernel<<<16, 32, 0, stream>>>(eaW0, 256,  7, 16, fW0enc);
    SGNP_wprep_kernel<<<64, 32, 0, stream>>>(eaW1, 128, 256,  8, fW1enc);
    SGNP_wprep_kernel<<<16, 32, 0, stream>>>(eaW2,  64, 128,  4, fW2enc);
    for (int blk = 0; blk < NBLK; ++blk)
        SGNP_wprep_kernel<<<8, 32, 0, stream>>>(gW + (size_t)blk * H_ * H_,
                                                64, 64, 4,
                                                fgW + (size_t)blk * 8 * 512);
    SGNP_wprep_kernel<<<32, 32, 0, stream>>>(hW0, 256,  64, 16, fhW0);
    SGNP_wprep_kernel<<<32, 32, 0, stream>>>(hW1,  64, 256,  4, fhW1);

    // ---- encoder: 2048 tiles of 16 nodes, 2 waves/block ----
    SGNP_enc_kernel<<<NNODE / 32, 64, 0, stream>>>(
        s_ctx, f_ctx, s_test, emb, fW0enc, eab0, fW1enc, eab1, fW2enc, eab2,
        lns, lnb, nodes, nodes16);

    // ---- knn (ctx + test queries) ----
    SGNP_knn_kernel<<<128, 256, 0, stream>>>(
        s_ctx, s_test, vlen, send, emask, edges);

    // ---- GAT blocks ----
    for (int blk = 0; blk < NBLK; ++blk) {
        SGNP_gat_h_kernel<<<NNODE / 32, 64, 0, stream>>>(
            nodes16, fgW + (size_t)blk * 8 * 512, gar + blk * H_, hbuf, hr);
        SGNP_gat_agg_kernel<<<NNODE / 4, 128, 0, stream>>>(
            nodes, nodes16, hbuf, hr, send, emask, edges,
            gas + blk * H_, gae + blk * D_, glns + blk * H_, glnb + blk * H_);
    }

    // ---- head on test nodes: 1024 tiles, 2 waves/block ----
    SGNP_head_kernel<<<BNT / 32, 64, 0, stream>>>(
        nodes16, fhW0, hb0, fhW1, hb1, hW2, hb2, out);
}
